// TrainableVarRegBlock_67765993996496
// MI455X (gfx1250) — matrix operand order repaired and checked by reading, not compiled, not run on hardware
//
#include <hip/hip_runtime.h>
#include <hip/hip_bf16.h>
#include <cmath>

typedef __attribute__((ext_vector_type(16))) _Float16 v16h;
typedef __attribute__((ext_vector_type(8)))  _Float16 v8h;
typedef __attribute__((ext_vector_type(8)))  float    v8f;

#define DVOL 96
#define NVOL (96*96*96)           // 884736
#define N48  (48*48*48)           // 110592
#define PD   98                   // padded dim (1-voxel halo)
#define PHW  (98*98)
#define NPAD (98*98*98)           // 941192 padded voxels

// ---------------------------------------------------------------------------
// Elementwise / resampling kernels (bandwidth-bound demons phase)
// ---------------------------------------------------------------------------

__global__ void resize_tri_kernel(const float* __restrict__ in, float* __restrict__ out,
                                  int C, int Di, int Hi, int Wi, int Do, int Ho, int Wo,
                                  float mul)
{
    long gid = (long)blockIdx.x * blockDim.x + threadIdx.x;
    long No = (long)Do * Ho * Wo;
    long total = (long)C * No;
    if (gid >= total) return;
    int c = (int)(gid / No);
    long idx = gid % No;
    int z = (int)(idx / (Ho * Wo));
    int y = (int)((idx / Wo) % Ho);
    int x = (int)(idx % Wo);
    float sz = (float)Di / Do, sy = (float)Hi / Ho, sx = (float)Wi / Wo;
    float cz = fminf(fmaxf((z + 0.5f) * sz - 0.5f, 0.f), (float)(Di - 1));
    float cy = fminf(fmaxf((y + 0.5f) * sy - 0.5f, 0.f), (float)(Hi - 1));
    float cx = fminf(fmaxf((x + 0.5f) * sx - 0.5f, 0.f), (float)(Wi - 1));
    int z0 = (int)floorf(cz), y0 = (int)floorf(cy), x0 = (int)floorf(cx);
    int z1 = min(z0 + 1, Di - 1), y1 = min(y0 + 1, Hi - 1), x1 = min(x0 + 1, Wi - 1);
    float fz = cz - z0, fy = cy - y0, fx = cx - x0;
    const float* p = in + (long)c * Di * Hi * Wi;
    long HWi = (long)Hi * Wi;
    float v000 = p[(long)z0*HWi + (long)y0*Wi + x0];
    float v001 = p[(long)z0*HWi + (long)y0*Wi + x1];
    float v010 = p[(long)z0*HWi + (long)y1*Wi + x0];
    float v011 = p[(long)z0*HWi + (long)y1*Wi + x1];
    float v100 = p[(long)z1*HWi + (long)y0*Wi + x0];
    float v101 = p[(long)z1*HWi + (long)y0*Wi + x1];
    float v110 = p[(long)z1*HWi + (long)y1*Wi + x0];
    float v111 = p[(long)z1*HWi + (long)y1*Wi + x1];
    float v00 = v000 + fx * (v001 - v000);
    float v01 = v010 + fx * (v011 - v010);
    float v10 = v100 + fx * (v101 - v100);
    float v11 = v110 + fx * (v111 - v110);
    float v0 = v00 + fy * (v01 - v00);
    float v1 = v10 + fy * (v11 - v10);
    out[gid] = mul * (v0 + fz * (v1 - v0));
}

__global__ void resize_nearest_kernel(const float* __restrict__ in, float* __restrict__ out,
                                      int Di, int Hi, int Wi, int Do, int Ho, int Wo)
{
    long gid = (long)blockIdx.x * blockDim.x + threadIdx.x;
    long No = (long)Do * Ho * Wo;
    if (gid >= No) return;
    int z = (int)(gid / (Ho * Wo));
    int y = (int)((gid / Wo) % Ho);
    int x = (int)(gid % Wo);
    int zi = min(Di - 1, (int)((z + 0.5f) * (float)Di / Do));
    int yi = min(Hi - 1, (int)((y + 0.5f) * (float)Hi / Ho));
    int xi = min(Wi - 1, (int)((x + 0.5f) * (float)Wi / Wo));
    out[gid] = in[(long)zi * Hi * Wi + (long)yi * Wi + xi];
}

// trilinear warp with 'nearest' boundary (clamped coords); vf layout [3][D][H][W]
__global__ void warp_kernel(const float* __restrict__ img, const float* __restrict__ vf,
                            float* __restrict__ out, int D, int H, int W)
{
    long Nl = (long)D * H * W;
    long gid = (long)blockIdx.x * blockDim.x + threadIdx.x;
    if (gid >= Nl) return;
    int z = (int)(gid / (H * W));
    int y = (int)((gid / W) % H);
    int x = (int)(gid % W);
    float cz = fminf(fmaxf((float)z + vf[gid],          0.f), (float)(D - 1));
    float cy = fminf(fmaxf((float)y + vf[Nl + gid],     0.f), (float)(H - 1));
    float cx = fminf(fmaxf((float)x + vf[2 * Nl + gid], 0.f), (float)(W - 1));
    int z0 = (int)floorf(cz), y0 = (int)floorf(cy), x0 = (int)floorf(cx);
    int z1 = min(z0 + 1, D - 1), y1 = min(y0 + 1, H - 1), x1 = min(x0 + 1, W - 1);
    float fz = cz - z0, fy = cy - y0, fx = cx - x0;
    long HW = (long)H * W;
    float v000 = img[(long)z0*HW + (long)y0*W + x0];
    float v001 = img[(long)z0*HW + (long)y0*W + x1];
    float v010 = img[(long)z0*HW + (long)y1*W + x0];
    float v011 = img[(long)z0*HW + (long)y1*W + x1];
    float v100 = img[(long)z1*HW + (long)y0*W + x0];
    float v101 = img[(long)z1*HW + (long)y0*W + x1];
    float v110 = img[(long)z1*HW + (long)y1*W + x0];
    float v111 = img[(long)z1*HW + (long)y1*W + x1];
    float v00 = v000 + fx * (v001 - v000);
    float v01 = v010 + fx * (v011 - v010);
    float v10 = v100 + fx * (v101 - v100);
    float v11 = v110 + fx * (v111 - v110);
    float v0 = v00 + fy * (v01 - v00);
    float v1 = v10 + fy * (v11 - v10);
    out[gid] = v0 + fz * (v1 - v0);
}

// active demons force + vf update (fused). jnp.gradient semantics.
__global__ void demon_update_kernel(const float* __restrict__ wrp, const float* __restrict__ fx,
                                    const float* __restrict__ msk, const float* __restrict__ spacing,
                                    float tau, float* __restrict__ vf, int D, int H, int W)
{
    long Nl = (long)D * H * W;
    long gid = (long)blockIdx.x * blockDim.x + threadIdx.x;
    if (gid >= Nl) return;
    int z = (int)(gid / (H * W));
    int y = (int)((gid / W) % H);
    int x = (int)(gid % W);
    long HW = (long)H * W;
    float c = wrp[gid];
    float gz = (z == 0)     ? wrp[gid + HW] - c :
               (z == D - 1) ? c - wrp[gid - HW] :
                              0.5f * (wrp[gid + HW] - wrp[gid - HW]);
    float gy = (y == 0)     ? wrp[gid + W] - c :
               (y == H - 1) ? c - wrp[gid - W] :
                              0.5f * (wrp[gid + W] - wrp[gid - W]);
    float gx = (x == 0)     ? wrp[gid + 1] - c :
               (x == W - 1) ? c - wrp[gid - 1] :
                              0.5f * (wrp[gid + 1] - wrp[gid - 1]);
    gz /= spacing[0]; gy /= spacing[1]; gx /= spacing[2];
    float diff = c - fx[gid];
    float denom = gz * gz + gy * gy + gx * gx + diff * diff;
    float f = (denom > 1e-9f) ? diff / denom : 0.f;
    float s = tau * msk[gid] * f;
    vf[gid]          += s * gz;
    vf[Nl + gid]     += s * gy;
    vf[2 * Nl + gid] += s * gx;
}

// 7-tap separable Gaussian along one axis, zero ('SAME') padding.
__global__ void gauss7_kernel(const float* __restrict__ in, float* __restrict__ out,
                              long total, long Nper, int S, int L,
                              float w0, float w1, float w2, float w3)
{
    long gid = (long)blockIdx.x * blockDim.x + threadIdx.x;
    if (gid >= total) return;
    long idx = gid % Nper;
    int c = (int)((idx / S) % L);
    float acc = w0 * in[gid];
    if (c >= 1)     acc += w1 * in[gid - (long)S];
    if (c + 1 < L)  acc += w1 * in[gid + (long)S];
    if (c >= 2)     acc += w2 * in[gid - 2L * S];
    if (c + 2 < L)  acc += w2 * in[gid + 2L * S];
    if (c >= 3)     acc += w3 * in[gid - 3L * S];
    if (c + 3 < L)  acc += w3 * in[gid + 3L * S];
    out[gid] = acc;
}

// pack CNN input: padded (98^3), channels-last (8: vfz,vfy,vfx, img*mask, warped, 0,0,0)
__global__ void pack_xin_kernel(const float* __restrict__ vf, const float* __restrict__ image,
                                const float* __restrict__ mask, const float* __restrict__ wrp,
                                _Float16* __restrict__ xin)
{
    long gid = (long)blockIdx.x * blockDim.x + threadIdx.x;
    if (gid >= NVOL) return;
    int z = (int)(gid / (DVOL * DVOL));
    int y = (int)((gid / DVOL) % DVOL);
    int x = (int)(gid % DVOL);
    _Float16* o = xin + (size_t)(((z + 1) * PD + (y + 1)) * PD + (x + 1)) * 8;
    o[0] = (_Float16)vf[gid];
    o[1] = (_Float16)vf[(long)NVOL + gid];
    o[2] = (_Float16)vf[2L * NVOL + gid];
    o[3] = (_Float16)(image[gid] * mask[gid]);
    o[4] = (_Float16)wrp[gid];
    o[5] = (_Float16)0.f; o[6] = (_Float16)0.f; o[7] = (_Float16)0.f;
}

__global__ void add3_kernel(float* __restrict__ dst, const float* __restrict__ a, long total)
{
    long gid = (long)blockIdx.x * blockDim.x + threadIdx.x;
    if (gid >= total) return;
    dst[gid] += a[gid];
}

// ---------------------------------------------------------------------------
// Instance norm (two pass) + Mish, on padded channels-last tensors
// ---------------------------------------------------------------------------

template <int C, int LOG2C>
__global__ __launch_bounds__(256)
void in_stats_kernel(const _Float16* __restrict__ a, float* __restrict__ stats)
{
    // blockDim=256 is a multiple of C -> each thread owns one fixed channel
    const int c = threadIdx.x & (C - 1);
    float s = 0.f, s2 = 0.f;
    const long total = (long)NVOL * C;
    for (long gid = (long)blockIdx.x * 256 + threadIdx.x; gid < total;
         gid += (long)gridDim.x * 256) {
        long vox = gid >> LOG2C;
        int z = (int)(vox / (DVOL * DVOL));
        int y = (int)((vox / DVOL) % DVOL);
        int x = (int)(vox % DVOL);
        float v = (float)a[(size_t)(((z + 1) * PD + (y + 1)) * PD + (x + 1)) * C + c];
        s += v; s2 += v * v;
    }
    __shared__ float sh[256], sh2[256];
    sh[threadIdx.x] = s; sh2[threadIdx.x] = s2;
    __syncthreads();
    for (int o = 128; o >= C; o >>= 1) {
        if ((int)threadIdx.x < o) { sh[threadIdx.x] += sh[threadIdx.x + o];
                                    sh2[threadIdx.x] += sh2[threadIdx.x + o]; }
        __syncthreads();
    }
    if ((int)threadIdx.x < C) {
        atomicAdd(&stats[2 * threadIdx.x],     sh[threadIdx.x]);
        atomicAdd(&stats[2 * threadIdx.x + 1], sh2[threadIdx.x]);
    }
}

template <int C, int LOG2C>
__global__ __launch_bounds__(256)
void norm_mish_kernel(_Float16* __restrict__ a, const float* __restrict__ stats)
{
    long gid = (long)blockIdx.x * 256 + threadIdx.x;
    long total = (long)NVOL * C;
    if (gid >= total) return;
    int c = (int)(gid & (C - 1));
    long vox = gid >> LOG2C;
    int z = (int)(vox / (DVOL * DVOL));
    int y = (int)((vox / DVOL) % DVOL);
    int x = (int)(vox % DVOL);
    size_t addr = (size_t)(((z + 1) * PD + (y + 1)) * PD + (x + 1)) * C + c;
    float mean = stats[2 * c] * (1.f / NVOL);
    float var  = stats[2 * c + 1] * (1.f / NVOL) - mean * mean;
    float v = ((float)a[addr] - mean) * rsqrtf(var + 1e-5f);
    float sp = (v > 20.f) ? v : log1pf(expf(v));
    a[addr] = (_Float16)(v * tanhf(sp));
}

// ---------------------------------------------------------------------------
// Implicit-GEMM conv3d 3x3x3 SAME via v_wmma_f32_16x16x32_f16.
// Activations: padded 98^3, channels-last (CINP = pow2 channel stride).
// K ordering: k = tap*CINP + ci  ->  each aligned 8-element A group is one
// contiguous b128 load; taps resolved via a 32-entry LDS offset table.
// B (weights) staged in LDS pre-shuffled into the per-lane WMMA layout.
// Two-stage ping-pong pipeline: prefetch A/B for step ks+1 into the other
// register buffer before the WMMA for step ks (hides load latency, removes
// WMMA->VALU hazard NOPs).
// ---------------------------------------------------------------------------

typedef union AU_u { v16h v; v8h g[2]; } au_t;

template <int CIN, int COUT, int CINP, int LOG2C, bool OUTF32>
__global__ __launch_bounds__(256)
void conv3d_wmma_kernel(const _Float16* __restrict__ act, const float* __restrict__ w,
                        const float* __restrict__ bias,
                        _Float16* __restrict__ outh, float* __restrict__ outf)
{
    constexpr int KTOTP  = 27 * CINP;
    constexpr int KSTEPS = (KTOTP + 31) / 32;
    __shared__ _Float16 bs[KSTEPS * 512];   // [kstep][lane][16 halfs], WMMA-B layout
    __shared__ int toff[32];

    const int cog = blockIdx.y;
    if (threadIdx.x < 32) {
        int t = threadIdx.x;
        toff[t] = (t < 27) ? ((t / 9) - 1) * PHW + (((t / 3) % 3) - 1) * PD + (t % 3 - 1) : 0;
    }
    for (int idx = threadIdx.x; idx < KSTEPS * 512; idx += 256) {
        int ks = idx >> 9;  int r = idx & 511;
        int lane = r >> 4;  int e = r & 15;
        int k  = ks * 32 + (e & 7) + ((e & 8) ? 16 : 0) + ((lane & 16) ? 8 : 0);
        int ci = k & (CINP - 1), tap = k >> LOG2C;
        int co = cog * 16 + (lane & 15);
        float v = 0.f;
        if (ci < CIN && tap < 27 && co < COUT) v = w[((long)co * CIN + ci) * 27 + tap];
        bs[idx] = (_Float16)v;
    }
    __syncthreads();

    const int lane = threadIdx.x & 31;
    const int wv   = threadIdx.x >> 5;
    const int tile = blockIdx.x * 8 + wv;        // NVOL/16 = 55296 = 6912*8 exact
    const int m    = lane & 15;
    const int h    = lane >> 4;                  // 0 or 1
    const int vbase = tile * 16;                 // 16 voxels share one (z,y) row: 96%16==0
    const int z  = vbase / (DVOL * DVOL);
    const int y  = (vbase / DVOL) % DVOL;
    const int x0 = vbase % DVOL;
    const int rowpad = ((z + 1) * PD + (y + 1)) * PD + (x0 + 1);
    const int apad   = rowpad + m;               // this lane's A voxel (padded idx)

    union { v8f v; float s[8]; } acc;
#pragma unroll
    for (int r = 0; r < 8; ++r) acc.s[r] = 0.f;

    au_t aP, aQ;
    v16h bP, bQ;
    auto loadA = [&](int ks, au_t& a) {
        const int k0 = ks * 32 + h * 8;
        const int k1 = k0 + 16;
        a.g[0] = *(const v8h*)(act + (size_t)(apad + toff[k0 >> LOG2C]) * CINP + (k0 & (CINP - 1)));
        a.g[1] = *(const v8h*)(act + (size_t)(apad + toff[k1 >> LOG2C]) * CINP + (k1 & (CINP - 1)));
    };
    auto loadB = [&](int ks, v16h& b) {
        b = *reinterpret_cast<const v16h*>(&bs[ks * 512 + lane * 16]);
    };

    loadA(0, aP); loadB(0, bP);
#pragma unroll
    for (int ks = 0; ks < KSTEPS; ks += 2) {
        if (ks + 1 < KSTEPS) { loadA(ks + 1, aQ); loadB(ks + 1, bQ); }
        acc.v = __builtin_amdgcn_wmma_f32_16x16x32_f16(
            false, aP.v, false, bP, (short)0, acc.v, false, false);
        if (ks + 1 < KSTEPS) {
            if (ks + 2 < KSTEPS) { loadA(ks + 2, aP); loadB(ks + 2, bP); }
            acc.v = __builtin_amdgcn_wmma_f32_16x16x32_f16(
                false, aQ.v, false, bQ, (short)0, acc.v, false, false);
        }
    }

    const int co = cog * 16 + (lane & 15);
#pragma unroll
    for (int r = 0; r < 8; ++r) {
        int M = r + h * 8;
        if (OUTF32) {
            if (co < COUT) outf[(size_t)co * NVOL + vbase + M] = acc.s[r] + bias[co];
        } else {
            outh[(size_t)(rowpad + M) * COUT + co] = (_Float16)acc.s[r];
        }
    }
}

// ---------------------------------------------------------------------------
// Host orchestration
// ---------------------------------------------------------------------------

static inline long divup(long a, long b) { return (a + b - 1) / b; }

extern "C" void kernel_launch(void* const* d_in, const int* in_sizes, int n_in,
                              void* d_out, int out_size, void* d_ws, size_t ws_size,
                              hipStream_t stream)
{
    (void)in_sizes; (void)n_in; (void)out_size; (void)ws_size;
    const float* image   = (const float*)d_in[0];
    const float* mask    = (const float*)d_in[1];
    const float* moving  = (const float*)d_in[2];
    const float* spacing = (const float*)d_in[3];
    const float* w1 = (const float*)d_in[4];
    const float* w2 = (const float*)d_in[5];
    const float* w3 = (const float*)d_in[6];
    const float* w4 = (const float*)d_in[7];
    const float* b4 = (const float*)d_in[8];
    float* out = (float*)d_out;

    // Gaussian 7-tap weights (sigma=1, radius=3, normalized)
    float e1 = expf(-0.5f), e2 = expf(-2.0f), e3 = expf(-4.5f);
    float ksum = 1.f + 2.f * (e1 + e2 + e3);
    float g0 = 1.f / ksum, g1 = e1 / ksum, g2 = e2 / ksum, g3 = e3 / ksum;

    // workspace layout
    char* p = (char*)d_ws;
    auto alloc = [&](size_t bytes) -> char* {
        char* r = p; p += (bytes + 255) & ~(size_t)255; return r;
    };
    float* f0    = (float*)alloc(N48 * 4);
    float* mv0   = (float*)alloc(N48 * 4);
    float* m0    = (float*)alloc(N48 * 4);
    float* w48   = (float*)alloc(N48 * 4);
    float* vf48  = (float*)alloc(3L * N48 * 4);
    float* tA48  = (float*)alloc(3L * N48 * 4);
    float* tB48  = (float*)alloc(3L * N48 * 4);
    float* vf    = (float*)alloc(3L * NVOL * 4);
    float* tA    = (float*)alloc(3L * NVOL * 4);
    float* tB    = (float*)alloc(3L * NVOL * 4);
    float* wrp   = (float*)alloc((long)NVOL * 4);
    float* corr  = (float*)alloc(3L * NVOL * 4);
    float* cvf   = (float*)alloc(3L * NVOL * 4);
    float* stats = (float*)alloc(128 * 4);
    _Float16* xin = (_Float16*)alloc(8L  * NPAD * 2);   // padded, channels-last
    _Float16* a1  = (_Float16*)alloc(32L * NPAD * 2);
    _Float16* a2  = (_Float16*)alloc(64L * NPAD * 2);

    const int T = 256;
    auto g1d = [&](long n) { return dim3((unsigned)divup(n, T)); };

    // ---------------- Level 0 : 48^3, tau=2, 2 iterations ----------------
    resize_tri_kernel<<<g1d(N48), T, 0, stream>>>(image,  f0,  1, 96,96,96, 48,48,48, 1.f);
    resize_tri_kernel<<<g1d(N48), T, 0, stream>>>(moving, mv0, 1, 96,96,96, 48,48,48, 1.f);
    resize_nearest_kernel<<<g1d(N48), T, 0, stream>>>(mask, m0, 96,96,96, 48,48,48);
    hipMemsetAsync(vf48, 0, 3L * N48 * 4, stream);
    for (int it = 0; it < 2; ++it) {
        warp_kernel<<<g1d(N48), T, 0, stream>>>(mv0, vf48, w48, 48, 48, 48);
        demon_update_kernel<<<g1d(N48), T, 0, stream>>>(w48, f0, m0, spacing, 2.0f, vf48, 48, 48, 48);
        gauss7_kernel<<<g1d(3L*N48), T, 0, stream>>>(vf48, tA48, 3L*N48, N48, 48*48, 48, g0,g1,g2,g3);
        gauss7_kernel<<<g1d(3L*N48), T, 0, stream>>>(tA48, tB48, 3L*N48, N48, 48,    48, g0,g1,g2,g3);
        gauss7_kernel<<<g1d(3L*N48), T, 0, stream>>>(tB48, vf48, 3L*N48, N48, 1,     48, g0,g1,g2,g3);
    }

    // match_vf 48^3 -> 96^3 (trilinear, magnitude x2 per axis)
    resize_tri_kernel<<<g1d(3L*NVOL), T, 0, stream>>>(vf48, vf, 3, 48,48,48, 96,96,96, 2.0f);

    // ---------------- Level 1 : 96^3, tau=2, 2 iterations ----------------
    for (int it = 0; it < 2; ++it) {
        warp_kernel<<<g1d(NVOL), T, 0, stream>>>(moving, vf, wrp, 96, 96, 96);
        demon_update_kernel<<<g1d(NVOL), T, 0, stream>>>(wrp, image, mask, spacing, 2.0f, vf, 96, 96, 96);
        gauss7_kernel<<<g1d(3L*NVOL), T, 0, stream>>>(vf, tA, 3L*NVOL, NVOL, 96*96, 96, g0,g1,g2,g3);
        gauss7_kernel<<<g1d(3L*NVOL), T, 0, stream>>>(tA, tB, 3L*NVOL, NVOL, 96,    96, g0,g1,g2,g3);
        gauss7_kernel<<<g1d(3L*NVOL), T, 0, stream>>>(tB, vf, 3L*NVOL, NVOL, 1,     96, g0,g1,g2,g3);
    }
    warp_kernel<<<g1d(NVOL), T, 0, stream>>>(moving, vf, wrp, 96, 96, 96);

    // ---------------- Correction CNN (WMMA f16, fp32 accumulate) ----------------
    // zero padded activation buffers (halo must be zero each call)
    hipMemsetAsync(xin, 0, 8L  * NPAD * 2, stream);
    hipMemsetAsync(a1,  0, 32L * NPAD * 2, stream);
    hipMemsetAsync(a2,  0, 64L * NPAD * 2, stream);
    pack_xin_kernel<<<g1d(NVOL), T, 0, stream>>>(vf, image, mask, wrp, xin);

    const dim3 cblk(256);
    // conv1: 5 -> 32 (CINP=8)
    conv3d_wmma_kernel<5, 32, 8, 3, false><<<dim3(6912, 2), cblk, 0, stream>>>(xin, w1, nullptr, a1, nullptr);
    hipMemsetAsync(stats, 0, 64 * 4, stream);
    in_stats_kernel<32, 5><<<dim3(1728), 256, 0, stream>>>(a1, stats);
    norm_mish_kernel<32, 5><<<g1d(32L*NVOL), 256, 0, stream>>>(a1, stats);
    // conv2: 32 -> 64
    conv3d_wmma_kernel<32, 64, 32, 5, false><<<dim3(6912, 4), cblk, 0, stream>>>(a1, w2, nullptr, a2, nullptr);
    hipMemsetAsync(stats, 0, 128 * 4, stream);
    in_stats_kernel<64, 6><<<dim3(1728), 256, 0, stream>>>(a2, stats);
    norm_mish_kernel<64, 6><<<g1d(64L*NVOL), 256, 0, stream>>>(a2, stats);
    // conv3: 64 -> 32 (reuse a1; interior fully overwritten, halo stays zero)
    conv3d_wmma_kernel<64, 32, 64, 6, false><<<dim3(6912, 2), cblk, 0, stream>>>(a2, w3, nullptr, a1, nullptr);
    hipMemsetAsync(stats, 0, 64 * 4, stream);
    in_stats_kernel<32, 5><<<dim3(1728), 256, 0, stream>>>(a1, stats);
    norm_mish_kernel<32, 5><<<g1d(32L*NVOL), 256, 0, stream>>>(a1, stats);
    // conv4: 32 -> 3, fp32 planar out + bias
    conv3d_wmma_kernel<32, 3, 32, 5, true><<<dim3(6912, 1), cblk, 0, stream>>>(a1, w4, b4, nullptr, corr);

    // corrected_vf = smooth3d(vf + correction)
    add3_kernel<<<g1d(3L*NVOL), T, 0, stream>>>(corr, vf, 3L*NVOL);
    gauss7_kernel<<<g1d(3L*NVOL), T, 0, stream>>>(corr, tA, 3L*NVOL, NVOL, 96*96, 96, g0,g1,g2,g3);
    gauss7_kernel<<<g1d(3L*NVOL), T, 0, stream>>>(tA,   tB, 3L*NVOL, NVOL, 96,    96, g0,g1,g2,g3);
    gauss7_kernel<<<g1d(3L*NVOL), T, 0, stream>>>(tB,  cvf, 3L*NVOL, NVOL, 1,     96, g0,g1,g2,g3);

    // outputs: [corrected_warped_moving (N), corrected_vf (3N), vf (3N)]
    warp_kernel<<<g1d(NVOL), T, 0, stream>>>(moving, cvf, out, 96, 96, 96);
    hipMemcpyAsync(out + (long)NVOL,       cvf, 3L * NVOL * 4, hipMemcpyDeviceToDevice, stream);
    hipMemcpyAsync(out + 4L * (long)NVOL,  vf,  3L * NVOL * 4, hipMemcpyDeviceToDevice, stream);
}
